// MultiheadSelfAttentionDec_48996986913026
// MI455X (gfx1250) — matrix operand order repaired and checked by reading, not compiled, not run on hardware
//
#include <hip/hip_runtime.h>
#include <hip/hip_bf16.h>
#include <math.h>

typedef __attribute__((ext_vector_type(16))) _Float16 v16h;
typedef __attribute__((ext_vector_type(8)))  _Float16 v8h;
typedef __attribute__((ext_vector_type(4)))  _Float16 v4h;
typedef __attribute__((ext_vector_type(8)))  float    v8f;

#define B_ 2
#define S_ 2048
#define C_ 2048
#define D_ 128
#define H_ 16
#define M_ (B_*S_)
#define EPS_ 1.1920928955078125e-07f

// ---------------------------------------------------------------- utilities
__global__ void f32_to_f16_kernel(const float* __restrict__ in,
                                  _Float16* __restrict__ out, int n) {
  int i = blockIdx.x * blockDim.x + threadIdx.x;
  int stride = gridDim.x * blockDim.x;
  for (; i < n; i += stride) out[i] = (_Float16)in[i];
}

// ------------------------------------------------------------- WMMA GEMM
// out[m,n] = sum_k A[m,k]*W[n,k] + bias[n]
// Block: 256 thr = 8 waves; block tile 256x128; wave tile 32x128.
// Each B fragment feeds 2 WMMAs (rows m0..15 and m0+16..31) -> halves
// weight-fetch per FLOP vs a 16-row wave tile.
template <bool F16OUT>
__global__ __launch_bounds__(256)
void gemm_wmma(const _Float16* __restrict__ A, const _Float16* __restrict__ W,
               const float* __restrict__ bias, float* __restrict__ outf,
               _Float16* __restrict__ outh, int Ndim, int Kdim) {
  const int lane = threadIdx.x & 31;
  const int wave = threadIdx.x >> 5;
  const int ln   = lane & 15;
  const int hi   = lane >> 4;
  const int n0   = blockIdx.x * 128;
  const int m0   = blockIdx.y * 256 + wave * 32;

  v8f z = {0.f,0.f,0.f,0.f,0.f,0.f,0.f,0.f};
  v8f acc[2][8];
#pragma unroll
  for (int f = 0; f < 2; ++f)
#pragma unroll
    for (int t = 0; t < 8; ++t) acc[f][t] = z;

  const _Float16* arow0 = A + (size_t)(m0 + ln) * Kdim;
  const _Float16* arow1 = A + (size_t)(m0 + 16 + ln) * Kdim;
  for (int k0 = 0; k0 < Kdim; k0 += 32) {
    // A fragment 16x32: elems 0..15 -> k0+16*hi.. (contiguous)
    v8h a0lo = *(const v8h*)(arow0 + k0 + 16 * hi);
    v8h a0hi = *(const v8h*)(arow0 + k0 + 16 * hi + 8);
    v16h a0  = __builtin_shufflevector(a0lo, a0hi, 0,1,2,3,4,5,6,7,8,9,10,11,12,13,14,15);
    v8h a1lo = *(const v8h*)(arow1 + k0 + 16 * hi);
    v8h a1hi = *(const v8h*)(arow1 + k0 + 16 * hi + 8);
    v16h a1  = __builtin_shufflevector(a1lo, a1hi, 0,1,2,3,4,5,6,7,8,9,10,11,12,13,14,15);
#pragma unroll
    for (int t = 0; t < 8; ++t) {
      // B fragment 32x16: lane = col n, 16 contiguous K halves at k0+16*hi
      const _Float16* wrow = W + (size_t)(n0 + t * 16 + ln) * Kdim + k0 + 16 * hi;
      v16h b = *(const v16h*)wrow;
      acc[0][t] = __builtin_amdgcn_wmma_f32_16x16x32_f16(
          false, a0, false, b, (short)0, acc[0][t], false, false);
      acc[1][t] = __builtin_amdgcn_wmma_f32_16x16x32_f16(
          false, a1, false, b, (short)0, acc[1][t], false, false);
    }
  }

#pragma unroll
  for (int t = 0; t < 8; ++t) {
    int n = n0 + t * 16 + ln;
    float bv = bias[n];
#pragma unroll
    for (int f = 0; f < 2; ++f) {
#pragma unroll
      for (int r = 0; r < 8; ++r) {
        int m = m0 + f * 16 + r + 8 * hi;   // C/D layout: m = r + 8*hi
        float v = acc[f][t][r] + bv;
        if (F16OUT) outh[(size_t)m * Ndim + n] = (_Float16)v;
        else        outf[(size_t)m * Ndim + n] = v;
      }
    }
  }
}

// ------------------------------------------- fused RMSNorm + rotary, relayout
// in:  [B,S,H*D] f16 (pre-norm projection), g:[D] f32, rope:[S,1,64,2,2] f32
// out: [B,H,S,D] f16
__global__ __launch_bounds__(256)
void rmsrope_kernel(const _Float16* __restrict__ in, const float* __restrict__ g,
                    const float* __restrict__ rope, _Float16* __restrict__ out) {
  const int lane = threadIdx.x & 31;
  int wid = blockIdx.x * (blockDim.x >> 5) + (threadIdx.x >> 5);  // over B*S*H
  int h = wid % H_;
  int s = (wid / H_) % S_;
  int b = wid / (H_ * S_);

  const int d = lane * 4;
  v4h xh = *(const v4h*)(in + (size_t)(b * S_ + s) * C_ + h * D_ + d);
  float x0 = (float)xh[0], x1 = (float)xh[1], x2 = (float)xh[2], x3 = (float)xh[3];
  float ss = x0 * x0 + x1 * x1 + x2 * x2 + x3 * x3;
#pragma unroll
  for (int m = 16; m >= 1; m >>= 1) ss += __shfl_xor(ss, m, 32);
  float r = rsqrtf(ss * (1.f / (float)D_) + EPS_);
  float y0 = x0 * r * g[d + 0], y1 = x1 * r * g[d + 1];
  float y2 = x2 * r * g[d + 2], y3 = x3 * r * g[d + 3];

  // rope pair i = d/2 : out[2i+j] = rope[s,0,i,j,0]*x[2i] + rope[s,0,i,j,1]*x[2i+1]
  const float* rp = rope + (size_t)s * 256 + (d / 2) * 4;
  float o0 = rp[0] * y0 + rp[1] * y1;
  float o1 = rp[2] * y0 + rp[3] * y1;
  float o2 = rp[4] * y2 + rp[5] * y3;
  float o3 = rp[6] * y2 + rp[7] * y3;

  v4h oh;
  oh[0] = (_Float16)o0; oh[1] = (_Float16)o1;
  oh[2] = (_Float16)o2; oh[3] = (_Float16)o3;
  *(v4h*)(out + ((size_t)(b * H_ + h) * S_ + s) * D_ + d) = oh;
}

// ------------------------------------------------ V relayout: [B,S,H,D]->[B,H,D,S]
__global__ void vtrans_kernel(const _Float16* __restrict__ v,
                              _Float16* __restrict__ vt) {
  size_t total = (size_t)B_ * H_ * D_ * S_;
  size_t stride = (size_t)gridDim.x * blockDim.x;
  for (size_t idx = (size_t)blockIdx.x * blockDim.x + threadIdx.x; idx < total;
       idx += stride) {
    int s = (int)(idx % S_);
    size_t t = idx / S_;
    int d = (int)(t % D_);
    size_t t2 = t / D_;
    int h = (int)(t2 % H_);
    int b = (int)(t2 / H_);
    vt[idx] = v[(size_t)(b * S_ + s) * C_ + h * D_ + d];
  }
}

// -------------------------------------------------- flash attention (wave32)
// q,k: [B,H,S,D] f16 ; vt: [B,H,D,S] f16 ; o: [B,S,H*D] f16
// one wave handles 16 queries for one (b,h); online softmax over 64-key blocks
// (4 score tiles pre-combined before each shuffle tree -> amortized reductions)
__global__ __launch_bounds__(256)
void attn_kernel(const _Float16* __restrict__ q, const _Float16* __restrict__ k,
                 const _Float16* __restrict__ vt, _Float16* __restrict__ o) {
  __shared__ __align__(32) _Float16 lds[8 * 16 * 64];
  const int lane = threadIdx.x & 31;
  const int wave = threadIdx.x >> 5;
  const int ln = lane & 15;
  const int hi = lane >> 4;

  int gw = blockIdx.x * 8 + wave;
  int qt = gw % (S_ / 16);
  int h  = (gw / (S_ / 16)) % H_;
  int b  = gw / ((S_ / 16) * H_);
  const int q0 = qt * 16;

  const size_t bh = (size_t)(b * H_ + h);
  const _Float16* qbase = q + (bh * S_ + q0) * D_;
  const _Float16* kbase = k + bh * S_ * D_;
  const _Float16* vbase = vt + bh * (size_t)D_ * S_;
  _Float16* plds = lds + wave * 16 * 64;

  // resident Q A-fragments: 4 chunks of K=32 over D=128
  v16h aq[4];
#pragma unroll
  for (int c = 0; c < 4; ++c) {
    const _Float16* ar = qbase + (size_t)ln * D_ + c * 32;
    v8h lo = *(const v8h*)(ar + 16 * hi);
    v8h hi8 = *(const v8h*)(ar + 16 * hi + 8);
    aq[c] = __builtin_shufflevector(lo, hi8, 0,1,2,3,4,5,6,7,8,9,10,11,12,13,14,15);
  }

  v8f z = {0.f,0.f,0.f,0.f,0.f,0.f,0.f,0.f};
  v8f accO[8];
#pragma unroll
  for (int t = 0; t < 8; ++t) accO[t] = z;
  float mrow[8], lrow[8];
#pragma unroll
  for (int r = 0; r < 8; ++r) { mrow[r] = -INFINITY; lrow[r] = 0.f; }

  const float scale = 0.08838834764831845f;  // 1/sqrt(128)

  for (int kb = 0; kb < S_; kb += 64) {
    // scores: four 16x16 tiles over the 64-key block, accumulate over D
    v8f sc[4];
#pragma unroll
    for (int nt = 0; nt < 4; ++nt) sc[nt] = z;
#pragma unroll
    for (int c = 0; c < 4; ++c) {
#pragma unroll
      for (int nt = 0; nt < 4; ++nt) {
        v16h bk = *(const v16h*)(kbase + (size_t)(kb + nt * 16 + ln) * D_ +
                                 c * 32 + 16 * hi);
        sc[nt] = __builtin_amdgcn_wmma_f32_16x16x32_f16(
            false, aq[c], false, bk, (short)0, sc[nt], false, false);
      }
    }
    // online softmax, rows m = r + 8*hi spread over 16-lane half-waves
#pragma unroll
    for (int r = 0; r < 8; ++r) {
      float v0 = sc[0][r] * scale, v1 = sc[1][r] * scale;
      float v2 = sc[2][r] * scale, v3 = sc[3][r] * scale;
      float rm = fmaxf(fmaxf(v0, v1), fmaxf(v2, v3));
#pragma unroll
      for (int msk = 8; msk >= 1; msk >>= 1) rm = fmaxf(rm, __shfl_xor(rm, msk, 32));
      float newm = fmaxf(mrow[r], rm);
      float corr = __expf(mrow[r] - newm);
      float p0 = __expf(v0 - newm);
      float p1 = __expf(v1 - newm);
      float p2 = __expf(v2 - newm);
      float p3 = __expf(v3 - newm);
      float psum = (p0 + p1) + (p2 + p3);
#pragma unroll
      for (int msk = 8; msk >= 1; msk >>= 1) psum += __shfl_xor(psum, msk, 32);
      lrow[r] = lrow[r] * corr + psum;
      mrow[r] = newm;
#pragma unroll
      for (int t = 0; t < 8; ++t) accO[t][r] *= corr;
      _Float16* prow = plds + (r + 8 * hi) * 64 + ln;
      prow[0]  = (_Float16)p0;
      prow[16] = (_Float16)p1;
      prow[32] = (_Float16)p2;
      prow[48] = (_Float16)p3;
    }
    asm volatile("s_wait_dscnt 0" ::: "memory");  // wave-local LDS RAW fence
    // reshape P (D-layout) -> two A-fragments (16x32 each) via LDS
    v16h ap[2];
#pragma unroll
    for (int c2 = 0; c2 < 2; ++c2) {
      const _Float16* pr = plds + ln * 64 + c2 * 32;
      v8h plo = *(const v8h*)(pr + 16 * hi);
      v8h phi = *(const v8h*)(pr + 16 * hi + 8);
      ap[c2] = __builtin_shufflevector(plo, phi, 0,1,2,3,4,5,6,7,8,9,10,11,12,13,14,15);
    }
    // O += P(16x64) @ V(64x128)
#pragma unroll
    for (int t = 0; t < 8; ++t) {
#pragma unroll
      for (int c2 = 0; c2 < 2; ++c2) {
        v16h bv = *(const v16h*)(vbase + (size_t)(t * 16 + ln) * S_ + kb +
                                 c2 * 32 + 16 * hi);
        accO[t] = __builtin_amdgcn_wmma_f32_16x16x32_f16(
            false, ap[c2], false, bv, (short)0, accO[t], false, false);
      }
    }
    asm volatile("s_wait_dscnt 0" ::: "memory");  // reads done before next stores
  }

#pragma unroll
  for (int r = 0; r < 8; ++r) {
    float inv = 1.f / lrow[r];
    int m = q0 + r + 8 * hi;
    size_t orow = (size_t)(b * S_ + m) * C_ + h * D_;
#pragma unroll
    for (int t = 0; t < 8; ++t)
      o[orow + t * 16 + ln] = (_Float16)(accO[t][r] * inv);
  }
}

// ---------------------------------------------------------------- launcher
extern "C" void kernel_launch(void* const* d_in, const int* in_sizes, int n_in,
                              void* d_out, int out_size, void* d_ws, size_t ws_size,
                              hipStream_t stream) {
  const float* x    = (const float*)d_in[0];
  const float* rope = (const float*)d_in[1];
  const float* Wq   = (const float*)d_in[2];
  const float* bq   = (const float*)d_in[3];
  const float* Wk   = (const float*)d_in[4];
  const float* bk   = (const float*)d_in[5];
  const float* Wv   = (const float*)d_in[6];
  const float* bv   = (const float*)d_in[7];
  const float* gq   = (const float*)d_in[8];
  const float* gk   = (const float*)d_in[9];
  const float* Wo   = (const float*)d_in[10];
  const float* bo   = (const float*)d_in[11];
  float* out = (float*)d_out;

  char* ws = (char*)d_ws;
  size_t off = 0;
  auto alloc = [&](size_t bytes) -> void* {
    void* p = ws + off;
    off += (bytes + 255) & ~(size_t)255;
    return p;
  };

  const size_t MC = (size_t)M_ * C_;   // 8M elements
  const size_t CC = (size_t)C_ * C_;   // 4M elements
  _Float16* xh   = (_Float16*)alloc(MC * 2);
  _Float16* wqh  = (_Float16*)alloc(CC * 2);
  _Float16* wkh  = (_Float16*)alloc(CC * 2);
  _Float16* wvh  = (_Float16*)alloc(CC * 2);
  _Float16* woh  = (_Float16*)alloc(CC * 2);
  _Float16* qpre = (_Float16*)alloc(MC * 2);   // also reused as attn output
  _Float16* kpre = (_Float16*)alloc(MC * 2);
  _Float16* vpre = (_Float16*)alloc(MC * 2);
  _Float16* qh   = (_Float16*)alloc(MC * 2);   // [B,H,S,D]
  _Float16* kh   = (_Float16*)alloc(MC * 2);   // [B,H,S,D]
  _Float16* vth  = (_Float16*)alloc(MC * 2);   // [B,H,D,S]
  _Float16* attn = qpre;                       // reuse after rmsrope consumes qpre

  // 1) precision conversion
  f32_to_f16_kernel<<<(int)((MC + 255) / 256), 256, 0, stream>>>(x, xh, (int)MC);
  f32_to_f16_kernel<<<(int)((CC + 255) / 256), 256, 0, stream>>>(Wq, wqh, (int)CC);
  f32_to_f16_kernel<<<(int)((CC + 255) / 256), 256, 0, stream>>>(Wk, wkh, (int)CC);
  f32_to_f16_kernel<<<(int)((CC + 255) / 256), 256, 0, stream>>>(Wv, wvh, (int)CC);
  f32_to_f16_kernel<<<(int)((CC + 255) / 256), 256, 0, stream>>>(Wo, woh, (int)CC);

  // 2) QKV projections (WMMA, f16 out); block tile 256x128
  dim3 ggrid(C_ / 128, M_ / 256);
  gemm_wmma<true><<<ggrid, 256, 0, stream>>>(xh, wqh, bq, nullptr, qpre, C_, C_);
  gemm_wmma<true><<<ggrid, 256, 0, stream>>>(xh, wkh, bk, nullptr, kpre, C_, C_);
  gemm_wmma<true><<<ggrid, 256, 0, stream>>>(xh, wvh, bv, nullptr, vpre, C_, C_);

  // 3) RMSNorm + rope + head relayout; V transpose
  int rr_blocks = (B_ * S_ * H_) / 8;
  rmsrope_kernel<<<rr_blocks, 256, 0, stream>>>(qpre, gq, rope, qh);
  rmsrope_kernel<<<rr_blocks, 256, 0, stream>>>(kpre, gk, rope, kh);
  vtrans_kernel<<<16384, 256, 0, stream>>>(vpre, vth);

  // 4) flash attention (WMMA, 64-key blocks)
  int attn_blocks = (B_ * H_ * (S_ / 16)) / 8;
  attn_kernel<<<attn_blocks, 256, 0, stream>>>(qh, kh, vth, attn);

  // 5) output projection (WMMA, f32 out + bias)
  gemm_wmma<false><<<ggrid, 256, 0, stream>>>(attn, woh, bo, out, nullptr, C_, C_);
}
